// SingleGNN_36240934043741
// MI455X (gfx1250) — matrix operand order repaired and checked by reading, not compiled, not run on hardware
//
#include <hip/hip_runtime.h>
#include <math.h>

// ---------------------------------------------------------------------------
// FeaStNet-style GNN forward for MI455X (gfx1250, wave32).
// GEMMs hoisted to node level, run on v_wmma_f32_16x16x32_f16 (f32 accum),
// with branch-free vectorized fragment loads (weights pre-transposed to f16,
// activations K-padded to multiples of 32, M handled by clamp+mask).
// Edge softmax-aggregation is gather/scatter bound -> VALU + f32 atomics.
// Head (32->1024->3) fused in one WMMA kernel through LDS.
// ---------------------------------------------------------------------------

#define HEADS 9
#define NEG_SLOPE 0.2f

typedef __attribute__((ext_vector_type(16))) _Float16 v16h;
typedef __attribute__((ext_vector_type(8)))  float    v8f;

__device__ __forceinline__ float leakyf(float v) { return v >= 0.f ? v : NEG_SLOPE * v; }

// ----------------------------- elementwise helpers -------------------------

__global__ void fill_kernel(float* __restrict__ p, float v, long n) {
    long i = (long)blockIdx.x * blockDim.x + threadIdx.x;
    if (i < n) p[i] = v;
}

// pad x (N x 6) -> xpad (N x 32), cols 6..31 zero
__global__ void pad_x_kernel(const float* __restrict__ x, float* __restrict__ xp, int N) {
    long i = (long)blockIdx.x * blockDim.x + threadIdx.x;
    if (i >= (long)N * 32) return;
    int n = (int)(i >> 5), c = (int)(i & 31);
    xp[i] = (c < 6) ? x[(size_t)n * 6 + c] : 0.f;
}

// transpose + convert weights: W (K x N, f32) -> Wt (N x Kp, f16), k >= K zero
__global__ void wt_kernel(const float* __restrict__ W, _Float16* __restrict__ Wt,
                          int K, int N, int Kp) {
    long i = (long)blockIdx.x * blockDim.x + threadIdx.x;
    if (i >= (long)N * Kp) return;
    int n = (int)(i / Kp), k = (int)(i % Kp);
    Wt[i] = (k < K) ? (_Float16)W[(size_t)k * N + n] : (_Float16)0.f;
}

__global__ void gather_rows_kernel(const float* __restrict__ x, const int* __restrict__ idx,
                                   float* __restrict__ out, int Nout, int C) {
    long i = (long)blockIdx.x * blockDim.x + threadIdx.x;
    if (i >= (long)Nout * C) return;
    int n = (int)(i / C), c = (int)(i % C);
    out[i] = x[(size_t)idx[n] * C + c];
}

__global__ void concat_kernel(const float* __restrict__ a, const float* __restrict__ b,
                              float* __restrict__ out, int N, int C1, int C2) {
    int C = C1 + C2;
    long i = (long)blockIdx.x * blockDim.x + threadIdx.x;
    if (i >= (long)N * C) return;
    int n = (int)(i / C), c = (int)(i % C);
    out[i] = (c < C1) ? a[(size_t)n * C1 + c] : b[(size_t)n * C2 + (c - C1)];
}

// segment_max via CAS float-max (max is order-independent -> deterministic)
__device__ __forceinline__ void atomicMaxF(float* addr, float val) {
    unsigned int* ua = (unsigned int*)addr;
    unsigned int old = *ua;
    while (__uint_as_float(old) < val) {
        unsigned int assumed = old;
        old = atomicCAS(ua, assumed, __float_as_uint(val));
        if (old == assumed) break;
    }
}

__global__ void seg_max_kernel(const float* __restrict__ x, const int* __restrict__ cluster,
                               float* __restrict__ out, int Nin, int C) {
    long i = (long)blockIdx.x * blockDim.x + threadIdx.x;
    if (i >= (long)Nin * C) return;
    int n = (int)(i / C), c = (int)(i % C);
    atomicMaxF(&out[(size_t)cluster[n] * C + c], x[i]);
}

// ----------------------------- WMMA GEMM -----------------------------------
// C[M,N] = A[M,Kp] @ Bt^T (+bias) (optional leaky).
//   A:  f32 row-major, row stride Kp (multiple of 32, rows 16B aligned)
//   Bt: f16, pre-transposed (N x Kp): Bt[n*Kp + k] = B[k,n]
// One wave per 16x16 C tile.  Branch-free: M edge handled by clamp + 0/1 mask.
// A fragment per lane: 16 contiguous floats (k0+hi*16 .. k0+hi*16+15).
// B fragment per lane: 16 contiguous f16 at column cn, k = k0+hi*16 (32B load).
__global__ void __launch_bounds__(32)
gemm_wmma_kernel(const float* __restrict__ A, const _Float16* __restrict__ Bt,
                 float* __restrict__ C, int M, int Kp, int N,
                 const float* __restrict__ bias, int do_leaky) {
    const int lane = threadIdx.x & 31;
    const int hi   = lane >> 4;
    const int l15  = lane & 15;
    const int row0 = blockIdx.y * 16;
    const int col0 = blockIdx.x * 16;
    const int r  = row0 + l15;
    const int cn = col0 + l15;
    const int rc = (r < M) ? r : (M - 1);
    const float msk = (r < M) ? 1.f : 0.f;
    const float*    Arow = A  + (size_t)rc * Kp;
    const _Float16* Bcol = Bt + (size_t)cn * Kp;
    v8f acc = {};
    for (int k0 = 0; k0 < Kp; k0 += 32) {
        const float4* ap = (const float4*)(Arow + k0 + hi * 16);
        float4 c0 = ap[0];
        float4 c1 = ap[1];
        const float4* ap2 = (const float4*)(Arow + k0 + hi * 16 + 8);
        float4 c2 = ap2[0];
        float4 c3 = ap2[1];
        __builtin_prefetch(Arow + k0 + 32 + hi * 8, 0, 1);  // global_prefetch_b8
        float af[16] = {c0.x, c0.y, c0.z, c0.w, c1.x, c1.y, c1.z, c1.w,
                        c2.x, c2.y, c2.z, c2.w, c3.x, c3.y, c3.z, c3.w};
        v16h a;
#pragma unroll
        for (int t = 0; t < 16; ++t) a[t] = (_Float16)(af[t] * msk);
        v16h b = *(const v16h*)(Bcol + k0 + hi * 16);
        acc = __builtin_amdgcn_wmma_f32_16x16x32_f16(false, a, false, b,
                                                     (short)0, acc, false, false);
    }
#pragma unroll
    for (int rr = 0; rr < 8; ++rr) {
        int m = row0 + rr + hi * 8;
        if (m < M) {
            float v = acc[rr];
            if (bias) v += bias[cn];
            if (do_leaky) v = leakyf(v);
            C[(size_t)m * N + cn] = v;
        }
    }
}

// ----------------------------- FEAST pieces --------------------------------

// xu[n,h] = sum_k x[n,k] * u[k,h]   (9 heads, tiny K -> scalar)
__global__ void node_xu_kernel(const float* __restrict__ x, const float* __restrict__ u,
                               float* __restrict__ xu, int N, int cin) {
    int n = blockIdx.x * blockDim.x + threadIdx.x;
    if (n >= N) return;
    float acc[HEADS];
#pragma unroll
    for (int h = 0; h < HEADS; ++h) acc[h] = 0.f;
    for (int k = 0; k < cin; ++k) {
        float xv = x[(size_t)n * cin + k];
        const float* ur = u + (size_t)k * HEADS;
#pragma unroll
        for (int h = 0; h < HEADS; ++h) acc[h] += xv * ur[h];
    }
#pragma unroll
    for (int h = 0; h < HEADS; ++h) xu[(size_t)n * HEADS + h] = acc[h];
}

__global__ void edge_deg_kernel(const int* __restrict__ d, int E0, float* __restrict__ deg) {
    int e = blockIdx.x * blockDim.x + threadIdx.x;
    if (e >= E0) return;
    atomicAdd(&deg[d[e]], 1.f);
}

// One wave per edge (incl. appended self-loops): per-lane softmax over 9 heads
// (redundant but cheap), lanes stride over out-channels -> coalesced xw gather
// and coalesced atomicAdd scatter into agg[dst].
__global__ void feast_edge_kernel(const float* __restrict__ xu, const float* __restrict__ xw,
                                  const float* __restrict__ cvec,
                                  const int* __restrict__ s, const int* __restrict__ d,
                                  int E0, int N, int cout, float* __restrict__ agg) {
    int wave = blockIdx.x * (blockDim.x >> 5) + (threadIdx.x >> 5);
    int lane = threadIdx.x & 31;
    int Et = E0 + N;
    if (wave >= Et) return;
    int se, de;
    if (wave < E0) { se = s[wave]; de = d[wave]; }
    else           { se = wave - E0; de = se; }
    float lg[HEADS];
    float mx = -3.4e38f;
#pragma unroll
    for (int h = 0; h < HEADS; ++h) {
        lg[h] = xu[(size_t)se * HEADS + h] - xu[(size_t)de * HEADS + h] + cvec[h];
        mx = fmaxf(mx, lg[h]);
    }
    float sum = 0.f;
#pragma unroll
    for (int h = 0; h < HEADS; ++h) { lg[h] = __expf(lg[h] - mx); sum += lg[h]; }
    float inv = 1.f / sum;
    const float* xr = xw + (size_t)se * HEADS * cout;
    float* ar = agg + (size_t)de * cout;
    for (int o = lane; o < cout; o += 32) {
        float m = 0.f;
#pragma unroll
        for (int h = 0; h < HEADS; ++h) m += lg[h] * xr[h * cout + o];
        atomicAdd(&ar[o], m * inv);
    }
}

__global__ void feast_finish_kernel(const float* __restrict__ agg, const float* __restrict__ deg,
                                    const float* __restrict__ b, float* __restrict__ dst,
                                    int N, int cout, int do_leaky) {
    long i = (long)blockIdx.x * blockDim.x + threadIdx.x;
    if (i >= (long)N * cout) return;
    int n = (int)(i / cout), o = (int)(i % cout);
    float v = agg[i] / fmaxf(deg[n], 1.f) + b[o];
    if (do_leaky) v = leakyf(v);
    dst[i] = v;
}

// ----------------------------- fused MLP head ------------------------------
// Per wave: 16 nodes.  hidden = leaky(feat(16x32) @ W1(32x1024) + b1) computed
// in 64 WMMA column tiles (W1 pre-transposed to f16), each tile staged in LDS
// and immediately contracted with W2(1024x3); then + b2 + x[:,3:6] residual
// and L2-normalize.  Hidden activation (245MB) never materialized.
__global__ void __launch_bounds__(32)
head_kernel(const float* __restrict__ feat, const _Float16* __restrict__ W1t,
            const float* __restrict__ b1, const float* __restrict__ W2,
            const float* __restrict__ b2, const float* __restrict__ x6,
            float* __restrict__ out, int N) {
    __shared__ float hbuf[16][17];
    __shared__ float oacc[16][3];
    const int lane = threadIdx.x & 31;
    const int hi = lane >> 4, l15 = lane & 15;
    const int m0 = blockIdx.x * 16;
    if (lane < 16) { oacc[lane][0] = b2[0]; oacc[lane][1] = b2[1]; oacc[lane][2] = b2[2]; }
    // A tile: feat rows m0..m0+15, Kp = 32 exactly (branch-free clamp+mask)
    const int r = m0 + l15;
    const int rc = (r < N) ? r : (N - 1);
    const float msk = (r < N) ? 1.f : 0.f;
    const float* Arow = feat + (size_t)rc * 32;
    float4 c0 = *(const float4*)(Arow + hi * 16);
    float4 c1 = *(const float4*)(Arow + hi * 16 + 4);
    float4 c2 = *(const float4*)(Arow + hi * 16 + 8);
    float4 c3 = *(const float4*)(Arow + hi * 16 + 12);
    float af[16] = {c0.x, c0.y, c0.z, c0.w, c1.x, c1.y, c1.z, c1.w,
                    c2.x, c2.y, c2.z, c2.w, c3.x, c3.y, c3.z, c3.w};
    v16h a;
#pragma unroll
    for (int t = 0; t < 16; ++t) a[t] = (_Float16)(af[t] * msk);
    __syncthreads();
    for (int j = 0; j < 64; ++j) {
        int col0 = j * 16;
        int cn = col0 + l15;
        v16h b = *(const v16h*)(W1t + (size_t)cn * 32 + hi * 16);
        v8f acc = {};
        acc = __builtin_amdgcn_wmma_f32_16x16x32_f16(false, a, false, b,
                                                     (short)0, acc, false, false);
        float bb = b1[cn];
#pragma unroll
        for (int rr = 0; rr < 8; ++rr)
            hbuf[rr + hi * 8][l15] = leakyf(acc[rr] + bb);
        __syncthreads();
        if (lane < 16) {
#pragma unroll
            for (int p = 0; p < 3; ++p) {
                float sacc = 0.f;
#pragma unroll
                for (int n2 = 0; n2 < 16; ++n2)
                    sacc += hbuf[lane][n2] * W2[(size_t)(col0 + n2) * 3 + p];
                oacc[lane][p] += sacc;
            }
        }
        __syncthreads();
    }
    if (lane < 16) {
        int m = m0 + lane;
        if (m < N) {
            float f0 = oacc[lane][0] + x6[(size_t)m * 6 + 3];
            float f1 = oacc[lane][1] + x6[(size_t)m * 6 + 4];
            float f2 = oacc[lane][2] + x6[(size_t)m * 6 + 5];
            float nrm = fmaxf(sqrtf(f0 * f0 + f1 * f1 + f2 * f2), 1e-12f);
            out[(size_t)m * 3 + 0] = f0 / nrm;
            out[(size_t)m * 3 + 1] = f1 / nrm;
            out[(size_t)m * 3 + 2] = f2 / nrm;
        }
    }
}

// ----------------------------- host side -----------------------------------

static inline int ceil_div(int a, int b) { return (a + b - 1) / b; }

static void fill_f(hipStream_t st, float* p, float v, long n) {
    fill_kernel<<<dim3((unsigned)((n + 255) / 256)), 256, 0, st>>>(p, v, n);
}

struct ConvP { const float *W, *u, *c, *b; };

// xg: GEMM input (row stride KpA, multiple of 32); xn: input for x@u (cin cols)
static void run_feast(hipStream_t st, const float* xg, int KpA, int Kreal,
                      const float* xn, int cin, int N, int cout,
                      const int* s, const int* d, int E0, const ConvP& P,
                      const float* deg, float* dest, int leaky,
                      float* xw, float* xu, float* agg, _Float16* wt) {
    int NC = HEADS * cout;
    wt_kernel<<<ceil_div(NC * KpA, 256), 256, 0, st>>>(P.W, wt, Kreal, NC, KpA);
    gemm_wmma_kernel<<<dim3(ceil_div(NC, 16), ceil_div(N, 16)), 32, 0, st>>>(
        xg, wt, xw, N, KpA, NC, nullptr, 0);
    node_xu_kernel<<<ceil_div(N, 256), 256, 0, st>>>(xn, P.u, xu, N, cin);
    fill_f(st, agg, 0.f, (long)N * cout);
    int Et = E0 + N;                     // + self loops
    feast_edge_kernel<<<ceil_div(Et, 8), 256, 0, st>>>(xu, xw, P.c, s, d, E0, N, cout, agg);
    feast_finish_kernel<<<ceil_div(N * cout, 256), 256, 0, st>>>(agg, deg, P.b, dest, N, cout, leaky);
}

// logical leaves: 0 x, 1 ei1, 2 c1, 3 ei2, 4 c2, 5 ei3,
// conv blocks (W,u,c,b) at 6+4*i for i in {l1,l2,l3,l4,r1,r2,r3,r4},
// 38 v1W, 39 v1b, 40 v2W, 41 v2b, 42 matW, 43 matb
static const long leaf_sizes[44] = {
    360000, 720000, 60000, 360000, 30000, 180000,
    1728, 54, 9, 32,          // l1 (6,32)
    18432, 288, 9, 64,        // l2 (32,64)
    73728, 576, 9, 128,       // l3 (64,128)
    147456, 1152, 9, 128,     // l4 (128,128)
    73728, 1152, 9, 64,       // r1 (128,64)
    73728, 1152, 9, 64,       // r2 (128,64)
    18432, 576, 9, 32,        // r3 (64,32)
    18432, 576, 9, 32,        // r4 (64,32)
    32768, 1024,              // fc_v1
    3072, 3,                  // fc_v2
    4096, 4                   // fc_mat (unused in forward)
};

static const int ORDER_A[44] = {  // pure insertion order
    0,1,2,3,4,5,
    6,7,8,9, 10,11,12,13, 14,15,16,17, 18,19,20,21,
    22,23,24,25, 26,27,28,29, 30,31,32,33, 34,35,36,37,
    38,39, 40,41, 42,43};
static const int ORDER_B[44] = {  // top insertion, params JAX-sorted (keys sorted)
    0,1,2,3,4,5,
    42,43, 38,39, 40,41,
    6,9,8,7, 10,13,12,11, 14,17,16,15, 18,21,20,19,
    22,25,24,23, 26,29,28,27, 30,33,32,31, 34,37,36,35};
static const int ORDER_C[44] = {  // fully JAX-sorted pytree
    2,4,1,3,5,
    42,43, 38,39, 40,41,
    6,9,8,7, 10,13,12,11, 14,17,16,15, 18,21,20,19,
    22,25,24,23, 26,29,28,27, 30,33,32,31, 34,37,36,35,
    0};

extern "C" void kernel_launch(void* const* d_in, const int* in_sizes, int n_in,
                              void* d_out, int out_size, void* d_ws, size_t ws_size,
                              hipStream_t stream) {
    if (n_in < 44 || out_size < 180000) return;
    const int* orders[3] = {ORDER_A, ORDER_B, ORDER_C};
    int chosen = 0;
    for (int c = 0; c < 3; ++c) {
        bool ok = true;
        for (int i = 0; i < 44; ++i)
            if ((long)in_sizes[i] != leaf_sizes[orders[c][i]]) { ok = false; break; }
        if (ok) { chosen = c; break; }
    }
    const int* ord = orders[chosen];
    const void* P[44];
    for (int i = 0; i < 44; ++i) P[ord[i]] = d_in[i];

    const int N1 = 60000, N2 = 30000, N3 = 15000;
    const int E1 = 360000, E2 = 180000, E3 = 90000;

    const float* x   = (const float*)P[0];
    const int*   ei1 = (const int*)P[1];
    const int*   cl1 = (const int*)P[2];
    const int*   ei2 = (const int*)P[3];
    const int*   cl2 = (const int*)P[4];
    const int*   ei3 = (const int*)P[5];
    const int *s1 = ei1, *d1 = ei1 + E1;
    const int *s2 = ei2, *d2 = ei2 + E2;
    const int *s3 = ei3, *d3 = ei3 + E3;

    ConvP cp[8];
    for (int i = 0; i < 8; ++i) {
        cp[i].W = (const float*)P[6 + 4 * i + 0];
        cp[i].u = (const float*)P[6 + 4 * i + 1];
        cp[i].c = (const float*)P[6 + 4 * i + 2];
        cp[i].b = (const float*)P[6 + 4 * i + 3];
    }
    const float* v1W = (const float*)P[38];
    const float* v1b = (const float*)P[39];
    const float* v2W = (const float*)P[40];
    const float* v2b = (const float*)P[41];

    // workspace layout (floats; every block a multiple of 8 floats -> 32B align)
    float* W = (float*)d_ws;
    size_t o = 0;
    float* xw   = W + o; o += 17280000;   // max N * 9*cout across all layers
    float* xu   = W + o; o += 540000;     // max N * 9
    float* agg  = W + o; o += 1920000;    // max N * cout
    float* deg1 = W + o; o += 60000;
    float* deg2 = W + o; o += 30000;
    float* deg3 = W + o; o += 15000;
    float* x1   = W + o; o += 1920000;    // N1 x 32
    float* x2   = W + o; o += 1920000;    // N2 x 64 (also pool-32 temp)
    float* x3   = W + o; o += 1920000;    // N3 x 128 (also pool-64 temp)
    float* f2   = W + o; o += 1920000;    // N2 x 64
    float* x2b  = W + o; o += 1920000;    // N2 x 64
    float* f1   = W + o; o += 1920000;    // N1 x 32
    float* feat = W + o; o += 1920000;    // N1 x 32
    float* up2  = W + o; o += 3840000;    // N2 x 128
    float* cat2 = W + o; o += 3840000;    // N2 x 128
    float* up1  = W + o; o += 3840000;    // N1 x 64
    float* cat1 = W + o; o += 3840000;    // N1 x 64
    float* xpad = W + o; o += 1920000;    // N1 x 32 (x zero-padded 6->32)
    _Float16* wt = (_Float16*)(W + o); o += 73728;  // max N*Kp f16 weight staging
    if (ws_size < o * sizeof(float)) return;

    hipStream_t st = stream;
    const float NEG_INF = -3.4e38f;

    // degrees (self loop contributes +1 at every node)
    fill_f(st, deg1, 1.f, N1);
    edge_deg_kernel<<<ceil_div(E1, 256), 256, 0, st>>>(d1, E1, deg1);
    fill_f(st, deg2, 1.f, N2);
    edge_deg_kernel<<<ceil_div(E2, 256), 256, 0, st>>>(d2, E2, deg2);
    fill_f(st, deg3, 1.f, N3);
    edge_deg_kernel<<<ceil_div(E3, 256), 256, 0, st>>>(d3, E3, deg3);

    // x1 = leaky(feast_l1(x))                        (N1, 32); GEMM uses padded x
    pad_x_kernel<<<ceil_div(N1 * 32, 256), 256, 0, st>>>(x, xpad, N1);
    run_feast(st, xpad, 32, 6, x, 6, N1, 32, s1, d1, E1, cp[0], deg1, x1, 1,
              xw, xu, agg, wt);

    // x2 = segment_max(x1, cluster1)                 (N2, 32)
    fill_f(st, x2, NEG_INF, (long)N2 * 32);
    seg_max_kernel<<<ceil_div(N1 * 32, 256), 256, 0, st>>>(x1, cl1, x2, N1, 32);
    // x2 = leaky(feast_l2(x2))                       (N2, 64)  (in-place safe)
    run_feast(st, x2, 32, 32, x2, 32, N2, 64, s2, d2, E2, cp[1], deg2, x2, 1,
              xw, xu, agg, wt);

    // x3 = segment_max(x2, cluster2)                 (N3, 64)
    fill_f(st, x3, NEG_INF, (long)N3 * 64);
    seg_max_kernel<<<ceil_div(N2 * 64, 256), 256, 0, st>>>(x2, cl2, x3, N2, 64);
    // x3 = leaky(feast_l3(x3)); x3 = leaky(feast_l4(x3))       (N3, 128)
    run_feast(st, x3, 64, 64, x3, 64, N3, 128, s3, d3, E3, cp[2], deg3, x3, 1,
              xw, xu, agg, wt);
    run_feast(st, x3, 128, 128, x3, 128, N3, 128, s3, d3, E3, cp[3], deg3, x3, 1,
              xw, xu, agg, wt);

    // f2 = feast_r1(x3[cluster2])                    (N2, 64), no leaky
    gather_rows_kernel<<<ceil_div(N2 * 128, 256), 256, 0, st>>>(x3, cl2, up2, N2, 128);
    run_feast(st, up2, 128, 128, up2, 128, N2, 64, s2, d2, E2, cp[4], deg2, f2, 0,
              xw, xu, agg, wt);

    // x2b = leaky(feast_r2(concat(x2, f2)))          (N2, 64)
    concat_kernel<<<ceil_div(N2 * 128, 256), 256, 0, st>>>(x2, f2, cat2, N2, 64, 64);
    run_feast(st, cat2, 128, 128, cat2, 128, N2, 64, s2, d2, E2, cp[5], deg2, x2b, 1,
              xw, xu, agg, wt);

    // f1 = feast_r3(x2b[cluster1])                   (N1, 32), no leaky
    gather_rows_kernel<<<ceil_div(N1 * 64, 256), 256, 0, st>>>(x2b, cl1, up1, N1, 64);
    run_feast(st, up1, 64, 64, up1, 64, N1, 32, s1, d1, E1, cp[6], deg1, f1, 0,
              xw, xu, agg, wt);

    // feat = leaky(feast_r4(concat(x1, f1)))         (N1, 32)
    concat_kernel<<<ceil_div(N1 * 64, 256), 256, 0, st>>>(x1, f1, cat1, N1, 32, 32);
    run_feast(st, cat1, 64, 64, cat1, 64, N1, 32, s1, d1, E1, cp[7], deg1, feat, 1,
              xw, xu, agg, wt);

    // fused head: leaky(feat@W1+b1)@W2 + b2 + x[:,3:6], then L2-normalize
    wt_kernel<<<ceil_div(1024 * 32, 256), 256, 0, st>>>(v1W, wt, 32, 1024, 32);
    head_kernel<<<ceil_div(N1, 16), 32, 0, st>>>(feat, wt, v1b, v2W, v2b, x,
                                                 (float*)d_out, N1);
}
